// MultiHeadAttention_41807211659875
// MI455X (gfx1250) — matrix operand order repaired and checked by reading, not compiled, not run on hardware
//
#include <hip/hip_runtime.h>
#include <hip/hip_bf16.h>

#define EMBED   1024
#define NHEADS  16
#define HDIM    64
#define BATCH   4
#define SEQ     2048
#define MTOT    (BATCH * SEQ)

typedef __bf16 bf16;
typedef __attribute__((ext_vector_type(4)))  __bf16 bf16x4;
typedef __attribute__((ext_vector_type(8)))  __bf16 bf16x8;
typedef __attribute__((ext_vector_type(16))) __bf16 v16bf;
typedef __attribute__((ext_vector_type(8)))  float  v8f;
typedef __attribute__((ext_vector_type(4)))  unsigned int u32x4;
typedef __attribute__((ext_vector_type(4)))  int i32x4;
typedef __attribute__((ext_vector_type(8)))  int i32x8;

// ---------------------------------------------------------------------------
// fp32 -> bf16 round-to-nearest-even
// ---------------------------------------------------------------------------
__device__ __forceinline__ bf16 f2bf(float f) {
  unsigned u = __builtin_bit_cast(unsigned, f);
  u += 0x7fffu + ((u >> 16) & 1u);
  unsigned short h = (unsigned short)(u >> 16);
  return __builtin_bit_cast(bf16, h);
}

// ---------------------------------------------------------------------------
// WMMA fragment loaders (CDNA5 16x16x32 bf16 layouts, wave32)
//
// A (16x32, MxK): lane L<16 holds M=L, elems 0..7 = K[hi*8 .. hi*8+7],
//                 elems 8..15 = K[16+hi*8 .. 16+hi*8+7]   (hi = lane>>4)
// B (32x16, KxN): lane holds N = lane&15, elems 0..7 = K[hi*8 .. hi*8+7],
//                 elems 8..15 = K[16+hi*8 .. 16+hi*8+7] (same as A)
// C/D (16x16 f32): slot r -> row = r + hi*8, col = lane&15
// ---------------------------------------------------------------------------
__device__ __forceinline__ v16bf load_frag_a(const bf16* __restrict__ src,
                                             int ld, int lane) {
  const int m  = lane & 15;
  const int hi = lane >> 4;
  const bf16* p = src + (size_t)m * ld + hi * 8;
  bf16x8 c0 = *(const bf16x8*)(p);
  bf16x8 c1 = *(const bf16x8*)(p + 16);
  return __builtin_shufflevector(c0, c1, 0, 1, 2, 3, 4, 5, 6, 7,
                                         8, 9, 10, 11, 12, 13, 14, 15);
}

__device__ __forceinline__ v16bf load_frag_b(const bf16* __restrict__ src,
                                             int ld, int lane) {
  const int n  = lane & 15;
  const int hi = lane >> 4;
  const bf16* p = src + (size_t)n * ld + hi * 8;
  bf16x8 c0 = *(const bf16x8*)(p);
  bf16x8 c1 = *(const bf16x8*)(p + 16);
  return __builtin_shufflevector(c0, c1, 0, 1, 2, 3, 4, 5, 6, 7,
                                         8, 9, 10, 11, 12, 13, 14, 15);
}

__device__ __forceinline__ v8f wmma_bf16(v16bf a, v16bf b, v8f c) {
  return __builtin_amdgcn_wmma_f32_16x16x32_bf16(
      /*neg_a=*/false, a, /*neg_b=*/false, b,
      /*c_mod=*/(short)0, c, /*reuse_a=*/false, /*reuse_b=*/false);
}

// ---------------------------------------------------------------------------
// TDM: async-DMA a [rows=128] x [cols=32] bf16 tile of a row-major [1024,1024]
// matrix into LDS (rows packed contiguously, 64B apart). Tracked by TENSORcnt.
// D# built per CDNA5 ISA 8.3/8.4: group0 = {flags, lds_addr, global_addr,
// type=2}; group1 = {data_size=2B, tensor dims, tile dims, dim0 stride}.
// ---------------------------------------------------------------------------
__device__ __forceinline__ void tdm_load_tile_128x32(const bf16* gsrc,
                                                     void* lds_dst) {
  const unsigned long long ga = (unsigned long long)(size_t)gsrc;
  u32x4 g0 = {};
  g0[0] = 1u;                                   // count=1 (valid), load mode
  g0[1] = (unsigned)(size_t)lds_dst;            // lds_addr (bytes)
  g0[2] = (unsigned)(ga & 0xffffffffu);         // global_addr[31:0]
  g0[3] = (unsigned)((ga >> 32) & 0x01ffffffu)  // global_addr[56:32]
          | (2u << 30);                         // type = 2 ("image")
  i32x8 g1 = {};
  g1[0] = 0x00010000;            // data_size=1 (2 bytes/elem), no multicast
  g1[1] = (int)(EMBED << 16);    // tensor_dim0[15:0]  (=1024)
  g1[2] = (int)(EMBED << 16);    // tensor_dim1[15:0]  (=1024)
  g1[3] = (int)(32 << 16);       // tile_dim0 = 32 elements
  g1[4] = 128;                   // tile_dim1 = 128 rows
  g1[5] = EMBED;                 // tensor_dim0_stride = 1024 elements
  i32x4 g2 = {};
  i32x4 g3 = {};
#if defined(__clang_major__) && (__clang_major__ >= 23)
  i32x8 g4 = {};
  __builtin_amdgcn_tensor_load_to_lds(g0, g1, g2, g3, g4, 0);
#else
  __builtin_amdgcn_tensor_load_to_lds(g0, g1, g2, g3, 0);
#endif
}

// ---------------------------------------------------------------------------
// fp32 -> bf16 conversion, 4 elems / thread
// ---------------------------------------------------------------------------
__global__ void cvt_f32_bf16(const float* __restrict__ s,
                             bf16* __restrict__ d, long n4) {
  long i = (long)blockIdx.x * blockDim.x + threadIdx.x;
  if (i >= n4) return;
  float4 f = ((const float4*)s)[i];
  bf16x4 o;
  o[0] = f2bf(f.x); o[1] = f2bf(f.y); o[2] = f2bf(f.z); o[3] = f2bf(f.w);
  ((bf16x4*)d)[i] = o;
}

// ---------------------------------------------------------------------------
// NT GEMM: out[m][o] = sum_k A[m][k] * W[o][k] + bias[o]
// A: bf16 [M,K] row-major, W: bf16 [O,K] row-major (torch Linear layout).
// Block = 256 threads = 8 waves (4 along M x 2 along N).
// Block tile 128x128; wave tile 32x64 (2x4 WMMA frags).
// W k-tiles (128x32 = 8 KB) are TDM-DMA'd into double-buffered LDS: issued
// for tile k+1 by wave 0 before waiting TENSORcnt on tile k.
// MODE 0: bf16 [M,N] out   (Q,K projections)
// MODE 1: f32  [M,N] out   (final output projection)
// MODE 2: bf16 head-transposed out -> Vt[((n*H + h)*D + d)*S + s]
// ---------------------------------------------------------------------------
template <int MODE>
__global__ __launch_bounds__(256) void gemm_nt_wmma(
    const bf16* __restrict__ A, const bf16* __restrict__ W,
    const float* __restrict__ bias, void* __restrict__ out,
    int M, int N, int K) {
  __shared__ __align__(32) bf16 bsm[2][128][32];  // double-buffered W tile

  const int lane = threadIdx.x & 31;
  const int wave = threadIdx.x >> 5;
  const int wm = wave & 3, wn = wave >> 2;
  const int m0 = blockIdx.x * 128 + wm * 32;
  const int n0base = blockIdx.y * 128;
  const int n0 = n0base + wn * 64;

  const bf16* wtile = W + (size_t)n0base * K;  // 128 rows of W for this block

  if (wave == 0) tdm_load_tile_128x32(wtile, &bsm[0][0][0]);

  v8f c[2][4] = {};
  for (int k0 = 0; k0 < K; k0 += 32) {
    const int cur = (k0 >> 5) & 1;
    if (wave == 0) {
      if (k0 + 32 < K) {
        tdm_load_tile_128x32(wtile + (k0 + 32), &bsm[cur ^ 1][0][0]);
        __builtin_amdgcn_s_wait_tensorcnt((short)1);  // tile k0 resident
      } else {
        __builtin_amdgcn_s_wait_tensorcnt((short)0);
      }
    }
    __syncthreads();  // W tile visible to all waves

    v16bf a0 = load_frag_a(A + (size_t)m0 * K + k0, K, lane);
    v16bf a1 = load_frag_a(A + (size_t)(m0 + 16) * K + k0, K, lane);
#pragma unroll
    for (int j = 0; j < 4; ++j) {
      v16bf b = load_frag_b(&bsm[cur][wn * 64 + j * 16][0], 32, lane);
      c[0][j] = wmma_bf16(a0, b, c[0][j]);
      c[1][j] = wmma_bf16(a1, b, c[1][j]);
    }
    __syncthreads();  // consumers done before buffer is re-filled
  }

  const int cn = lane & 15;
  const int hi = lane >> 4;
  for (int j = 0; j < 4; ++j) {
    const int col = n0 + j * 16 + cn;
    const float bv = bias[col];
    for (int i = 0; i < 2; ++i) {
      for (int r = 0; r < 8; ++r) {
        const int row = m0 + i * 16 + r + hi * 8;
        const float v = c[i][j][r] + bv;
        if (MODE == 1) {
          ((float*)out)[(size_t)row * N + col] = v;
        } else if (MODE == 0) {
          ((bf16*)out)[(size_t)row * N + col] = f2bf(v);
        } else {  // MODE 2: per-head transpose for V
          const int n = row / SEQ, s = row % SEQ;
          const int h = col / HDIM, d = col % HDIM;
          ((bf16*)out)[(((size_t)n * NHEADS + h) * HDIM + d) * SEQ + s] =
              f2bf(v);
        }
      }
    }
  }
}

// ---------------------------------------------------------------------------
// Flash-attention: one block per (q-block of 64, n*H). 4 waves; each wave
// owns a 16-row Q tile, streams key blocks of 64, keeps running (m, l, O).
// Q,K: bf16 [N,S,E] (head slice via +h*HDIM, ld=E). Vt: bf16 [N,H,D,S].
// ctx out: bf16 [N,S,E].
// ---------------------------------------------------------------------------
__global__ __launch_bounds__(128) void attention_wmma(
    const bf16* __restrict__ Q, const bf16* __restrict__ Kmat,
    const bf16* __restrict__ Vt, bf16* __restrict__ ctx) {
  __shared__ bf16 pbuf[4][16][72];  // per-wave P tile (16x64, padded stride)

  const int lane = threadIdx.x & 31;
  const int wave = threadIdx.x >> 5;
  const int hi = lane >> 4;
  const int lr = lane & 15;
  const int h = blockIdx.y % NHEADS;
  const int n = blockIdx.y / NHEADS;
  const int q0 = blockIdx.x * 64 + wave * 16;
  const float scale = 0.125f;  // 1/sqrt(64)

  // Q fragments for this wave's 16 rows (K dim = HDIM = 64 -> 2 frags)
  const bf16* qbase = Q + ((size_t)n * SEQ + q0) * EMBED + h * HDIM;
  v16bf qa0 = load_frag_a(qbase, EMBED, lane);
  v16bf qa1 = load_frag_a(qbase + 32, EMBED, lane);

  float m[8], l[8];
  v8f acc[4] = {};
  for (int r = 0; r < 8; ++r) { m[r] = -1e30f; l[r] = 0.f; }

  const bf16* khead = Kmat + (size_t)n * SEQ * EMBED + h * HDIM;
  const bf16* vt_head = Vt + ((size_t)n * NHEADS + h) * HDIM * (size_t)SEQ;

  for (int key0 = 0; key0 < SEQ; key0 += 64) {
    // prefetch next key block's K rows and Vt columns (global_prefetch_b8)
    if (key0 + 64 < SEQ) {
      __builtin_prefetch(khead + (size_t)(key0 + 64 + lane * 2) * EMBED, 0, 3);
      __builtin_prefetch(vt_head + (size_t)(lane * 2) * SEQ + key0 + 64, 0, 3);
    }

    // ---- S = Q @ K^T over this 64-key block (4 n-tiles x 2 k-steps) ----
    const bf16* kbase = khead + (size_t)key0 * EMBED;
    v8f sfr[4];
#pragma unroll
    for (int nt = 0; nt < 4; ++nt) {
      v8f sa = {};
      const bf16* kb = kbase + (size_t)(nt * 16) * EMBED;
      sa = wmma_bf16(qa0, load_frag_b(kb, EMBED, lane), sa);
      sa = wmma_bf16(qa1, load_frag_b(kb + 32, EMBED, lane), sa);
      sfr[nt] = sa;
    }

    // ---- online softmax: row max across 4 tiles + 16 lanes/row ----
    float tm[8];
    for (int r = 0; r < 8; ++r) {
      float v = fmaxf(fmaxf(sfr[0][r], sfr[1][r]),
                      fmaxf(sfr[2][r], sfr[3][r]));
      tm[r] = v * scale;
    }
    for (int msk = 1; msk < 16; msk <<= 1)
      for (int r = 0; r < 8; ++r)
        tm[r] = fmaxf(tm[r], __shfl_xor(tm[r], msk, 32));

    float alpha[8];
    for (int r = 0; r < 8; ++r) {
      float nm = fmaxf(m[r], tm[r]);
      alpha[r] = __expf(m[r] - nm);
      m[r] = nm;
    }

    // ---- P = exp(scale*S - m); stash bf16 P in LDS (C -> A layout) ----
    float ts[8] = {0.f, 0.f, 0.f, 0.f, 0.f, 0.f, 0.f, 0.f};
    for (int nt = 0; nt < 4; ++nt) {
      for (int r = 0; r < 8; ++r) {
        float e = __expf(sfr[nt][r] * scale - m[r]);
        ts[r] += e;
        pbuf[wave][r + hi * 8][nt * 16 + lr] = f2bf(e);
      }
    }
    for (int msk = 1; msk < 16; msk <<= 1)
      for (int r = 0; r < 8; ++r) ts[r] += __shfl_xor(ts[r], msk, 32);
    for (int r = 0; r < 8; ++r) l[r] = l[r] * alpha[r] + ts[r];
    for (int nt = 0; nt < 4; ++nt)
      for (int r = 0; r < 8; ++r) acc[nt][r] *= alpha[r];

    // ---- O += P @ V  (A frags reloaded from LDS; B rows from Vt) ----
    const bf16* vtb = vt_head + key0;
#pragma unroll
    for (int ks = 0; ks < 2; ++ks) {
      v16bf pa = load_frag_a(&pbuf[wave][0][0] + ks * 32, 72, lane);
#pragma unroll
      for (int nt = 0; nt < 4; ++nt) {
        v16bf vb = load_frag_b(vtb + (size_t)(nt * 16) * SEQ + ks * 32,
                               SEQ, lane);
        acc[nt] = wmma_bf16(pa, vb, acc[nt]);
      }
    }
  }

  // ---- epilogue: O /= l, write bf16 context [N,S,E] ----
  bf16* obase = ctx + ((size_t)n * SEQ + q0) * EMBED + h * HDIM;
  for (int r = 0; r < 8; ++r) {
    const float inv = 1.0f / l[r];
    const int row = r + hi * 8;
    for (int nt = 0; nt < 4; ++nt)
      obase[(size_t)row * EMBED + nt * 16 + lr] = f2bf(acc[nt][r] * inv);
  }
}

// ---------------------------------------------------------------------------
// Host launcher
// ---------------------------------------------------------------------------
extern "C" void kernel_launch(void* const* d_in, const int* in_sizes, int n_in,
                              void* d_out, int out_size, void* d_ws,
                              size_t ws_size, hipStream_t stream) {
  const float* x  = (const float*)d_in[0];
  const float* Wq = (const float*)d_in[1];
  const float* bq = (const float*)d_in[2];
  const float* Wk = (const float*)d_in[3];
  const float* bk = (const float*)d_in[4];
  const float* Wv = (const float*)d_in[5];
  const float* bv = (const float*)d_in[6];
  const float* Wo = (const float*)d_in[7];
  const float* bo = (const float*)d_in[8];
  float* out = (float*)d_out;

  const size_t XE = (size_t)MTOT * EMBED;   // 8192*1024
  const size_t WE = (size_t)EMBED * EMBED;  // 1024*1024

  char* ws = (char*)d_ws;
  size_t off = 0;
  bf16* xb  = (bf16*)(ws + off); off += XE * sizeof(bf16);  // reused as ctx
  bf16* wqb = (bf16*)(ws + off); off += WE * sizeof(bf16);
  bf16* wkb = (bf16*)(ws + off); off += WE * sizeof(bf16);
  bf16* wvb = (bf16*)(ws + off); off += WE * sizeof(bf16);
  bf16* wob = (bf16*)(ws + off); off += WE * sizeof(bf16);
  bf16* qb  = (bf16*)(ws + off); off += XE * sizeof(bf16);
  bf16* kb  = (bf16*)(ws + off); off += XE * sizeof(bf16);
  bf16* vtb = (bf16*)(ws + off); off += XE * sizeof(bf16);
  bf16* ctx = xb;  // x_bf16 is dead after the QKV projections

  // 1) fp32 -> bf16 conversions
  {
    long n4 = (long)(XE / 4);
    cvt_f32_bf16<<<(unsigned)((n4 + 255) / 256), 256, 0, stream>>>(x, xb, n4);
    long w4 = (long)(WE / 4);
    unsigned wg = (unsigned)((w4 + 255) / 256);
    cvt_f32_bf16<<<wg, 256, 0, stream>>>(Wq, wqb, w4);
    cvt_f32_bf16<<<wg, 256, 0, stream>>>(Wk, wkb, w4);
    cvt_f32_bf16<<<wg, 256, 0, stream>>>(Wv, wvb, w4);
    cvt_f32_bf16<<<wg, 256, 0, stream>>>(Wo, wob, w4);
  }

  // 2) QKV projections (M=8192, N=1024, K=1024), 128x128 block tiles
  dim3 ggrid(MTOT / 128, EMBED / 128);
  gemm_nt_wmma<0><<<ggrid, 256, 0, stream>>>(xb, wqb, bq, qb,
                                             MTOT, EMBED, EMBED);
  gemm_nt_wmma<0><<<ggrid, 256, 0, stream>>>(xb, wkb, bk, kb,
                                             MTOT, EMBED, EMBED);
  gemm_nt_wmma<2><<<ggrid, 256, 0, stream>>>(xb, wvb, bv, vtb,
                                             MTOT, EMBED, EMBED);

  // 3) flash attention -> ctx (bf16 [N,S,E])
  dim3 agrid(SEQ / 64, BATCH * NHEADS);
  attention_wmma<<<agrid, 128, 0, stream>>>(qb, kb, vtb, ctx);

  // 4) output projection -> f32 d_out
  gemm_nt_wmma<1><<<ggrid, 256, 0, stream>>>(ctx, wob, bo, out,
                                             MTOT, EMBED, EMBED);

  (void)in_sizes; (void)n_in; (void)out_size; (void)ws_size;
}
